// AttentionPooling_75265006895955
// MI455X (gfx1250) — matrix operand order repaired and checked by reading, not compiled, not run on hardware
//
#include <hip/hip_runtime.h>
#include <hip/hip_bf16.h>

#define HIDDEN 768
#define BATCH  4
#define SEQ    8192
#define NQ     1024

typedef __attribute__((ext_vector_type(16))) __bf16 v16bf;
typedef __attribute__((ext_vector_type(8)))  float  v8f;

// ---- helpers -------------------------------------------------------------
// Hardware bf16 conversion (fptrunc, RNE) — lowers to v_cvt_pk_bf16_f32
static __device__ __forceinline__ unsigned short f32_bf16(float f) {
  union { __bf16 b; unsigned short u; } c; c.b = (__bf16)f; return c.u;
}
static __device__ __forceinline__ unsigned pack2_bf16(float lo, float hi) {
  union { __bf16 b[2]; unsigned u; } c;
  c.b[0] = (__bf16)lo; c.b[1] = (__bf16)hi; return c.u;
}

// A-matrix (16x32 bf16) per-lane data: k = half*8 + [0..7] and 16 + half*8 + [0..7]
static __device__ __forceinline__ v16bf loadFragA(const unsigned short* base, int half) {
  union { v16bf v; uint4 q[2]; } f;
  f.q[0] = *reinterpret_cast<const uint4*>(base + (half << 3));
  f.q[1] = *reinterpret_cast<const uint4*>(base + 16 + (half << 3));
  return f.v;
}
// B-matrix (32x16 bf16) per-lane data: k = half*8 + [0..7] and 16 + half*8 + [0..7]
static __device__ __forceinline__ v16bf loadFragB(const unsigned short* base, int half) {
  union { v16bf v; uint4 q[2]; } f;
  f.q[0] = *reinterpret_cast<const uint4*>(base + (half << 3));
  f.q[1] = *reinterpret_cast<const uint4*>(base + 16 + (half << 3));
  return f.v;
}
static __device__ __forceinline__ v8f wmma_bf16(v16bf a, v16bf b, v8f c) {
  return __builtin_amdgcn_wmma_f32_16x16x32_bf16(false, a, false, b, (short)0, c, false, false);
}

// ---- Kernel 1: bf16(X[M,768] @ W[768,768] + bias) ------------------------
// block = 256 threads (8 waves), block tile 64x64, wave tile 16x32.
// Ping-pong LDS double buffering: one barrier per K-step.
// transT = 0: out[row][col] row-major [M,768]
// transT = 1: out[b][col][s] (i.e. V^T), row = b*SEQ + s

__global__ __launch_bounds__(256) void proj_gemm_bf16(
    const float* __restrict__ X, const float* __restrict__ W,
    const float* __restrict__ bias, unsigned short* __restrict__ out,
    const int transT)
{
  __shared__ __align__(16) unsigned short At[2][64 * 32];  // [m][k] bf16
  __shared__ __align__(16) unsigned short Bt[2][64 * 32];  // [n][k] bf16 (transposed)
  const int tid  = threadIdx.x;
  const int lane = tid & 31, wave = tid >> 5;
  const int half = lane >> 4, r = lane & 15;
  const int m0 = blockIdx.x * 64, n0 = blockIdx.y * 64;
  const int wm = (wave & 3) * 16;          // wave row offset in block tile
  const int wn = (wave >> 2) * 32;         // wave col offset in block tile
  // per-thread staging coordinates
  const int i0 = tid * 8;                  // A: 8 contiguous k of one row
  const int mr = i0 >> 5, kc = i0 & 31;
  const int kp = (tid >> 4) * 2;           // B: k rows kp, kp+1
  const int nc = (tid & 15) * 4;           // B: 4 n cols

  const float* aPtr  = &X[(size_t)(m0 + mr) * HIDDEN + kc];
  const float* bPtr0 = &W[(size_t)kp * HIDDEN + n0 + nc];
  const float* bPtr1 = bPtr0 + HIDDEN;

  float4 xa0, xa1, w0, w1;
  xa0 = *reinterpret_cast<const float4*>(aPtr);
  xa1 = *reinterpret_cast<const float4*>(aPtr + 4);
  w0  = *reinterpret_cast<const float4*>(bPtr0);
  w1  = *reinterpret_cast<const float4*>(bPtr1);
  {
    uint4 pk;
    pk.x = pack2_bf16(xa0.x, xa0.y); pk.y = pack2_bf16(xa0.z, xa0.w);
    pk.z = pack2_bf16(xa1.x, xa1.y); pk.w = pack2_bf16(xa1.z, xa1.w);
    *reinterpret_cast<uint4*>(&At[0][i0]) = pk;
    *reinterpret_cast<unsigned*>(&Bt[0][(nc + 0) * 32 + kp]) = pack2_bf16(w0.x, w1.x);
    *reinterpret_cast<unsigned*>(&Bt[0][(nc + 1) * 32 + kp]) = pack2_bf16(w0.y, w1.y);
    *reinterpret_cast<unsigned*>(&Bt[0][(nc + 2) * 32 + kp]) = pack2_bf16(w0.z, w1.z);
    *reinterpret_cast<unsigned*>(&Bt[0][(nc + 3) * 32 + kp]) = pack2_bf16(w0.w, w1.w);
  }
  __syncthreads();

  v8f c0 = {}, c1 = {};
  int buf = 0;
  for (int kk = 0; kk < HIDDEN; kk += 32, buf ^= 1) {
    const bool more = (kk + 32) < HIDDEN;
    if (more) {   // issue next tile's global loads before consuming current
      xa0 = *reinterpret_cast<const float4*>(aPtr + kk + 32);
      xa1 = *reinterpret_cast<const float4*>(aPtr + kk + 36);
      w0  = *reinterpret_cast<const float4*>(bPtr0 + (size_t)(kk + 32) * HIDDEN);
      w1  = *reinterpret_cast<const float4*>(bPtr1 + (size_t)(kk + 32) * HIDDEN);
    }
    const v16bf a  = loadFragA(&At[buf][(wm + r) * 32], half);
    const v16bf b0 = loadFragB(&Bt[buf][(wn + r) * 32], half);
    const v16bf b1 = loadFragB(&Bt[buf][(wn + 16 + r) * 32], half);
    c0 = wmma_bf16(a, b0, c0);
    c1 = wmma_bf16(a, b1, c1);
    if (more) {
      const int nb = buf ^ 1;
      uint4 pk;
      pk.x = pack2_bf16(xa0.x, xa0.y); pk.y = pack2_bf16(xa0.z, xa0.w);
      pk.z = pack2_bf16(xa1.x, xa1.y); pk.w = pack2_bf16(xa1.z, xa1.w);
      *reinterpret_cast<uint4*>(&At[nb][i0]) = pk;
      *reinterpret_cast<unsigned*>(&Bt[nb][(nc + 0) * 32 + kp]) = pack2_bf16(w0.x, w1.x);
      *reinterpret_cast<unsigned*>(&Bt[nb][(nc + 1) * 32 + kp]) = pack2_bf16(w0.y, w1.y);
      *reinterpret_cast<unsigned*>(&Bt[nb][(nc + 2) * 32 + kp]) = pack2_bf16(w0.z, w1.z);
      *reinterpret_cast<unsigned*>(&Bt[nb][(nc + 3) * 32 + kp]) = pack2_bf16(w0.w, w1.w);
      __syncthreads();
    }
  }

  const int col0 = n0 + wn + r, col1 = col0 + 16;
  const float bb0 = bias[col0], bb1 = bias[col1];
  if (!transT) {
    for (int g = 0; g < 8; ++g) {
      const int row = m0 + wm + g + half * 8;
      out[(size_t)row * HIDDEN + col0] = f32_bf16(c0[g] + bb0);
      out[(size_t)row * HIDDEN + col1] = f32_bf16(c1[g] + bb1);
    }
  } else {
    for (int g = 0; g < 8; ++g) {
      const int row = m0 + wm + g + half * 8;   // = b*SEQ + s
      const int bb = row >> 13, s = row & (SEQ - 1);
      out[((size_t)bb * HIDDEN + col0) * SEQ + s] = f32_bf16(c0[g] + bb0);
      out[((size_t)bb * HIDDEN + col1) * SEQ + s] = f32_bf16(c1[g] + bb1);
    }
  }
}

// ---- Kernel 2: flash attention, 256-key outer step ------------------------
// grid (Q/16, B); block = 256 threads (8 waves) owns 16 query rows x 768 cols.
// Per 256-key tile: each wave privately computes full-K scores for its own
// 32 keys (Q from LDS), softmax merges per-row max/sum across waves through
// two tiny LDS buffers, exp-tiles land in a shared 16x256 bf16 P buffer,
// then each wave runs P @ V for its 96 d-cols (V^T from global).
// 2 barriers and zero LDS atomics per 256 keys.

__global__ __launch_bounds__(256) void attn_flash_bf16(
    const unsigned short* __restrict__ qb, const unsigned short* __restrict__ kb,
    const unsigned short* __restrict__ vbT, const float* __restrict__ mask,
    float* __restrict__ out)
{
  __shared__ __align__(16) unsigned short qTile[16 * HIDDEN]; // 24 KB
  __shared__ __align__(16) unsigned short pBuf[16 * 256];     // 8 KB
  __shared__ __align__(16) float rowMaxB[16][8];
  __shared__ __align__(16) float rowSumB[16][8];
  const int tid  = threadIdx.x;
  const int lane = tid & 31, wave = tid >> 5;
  const int half = lane >> 4, r = lane & 15;
  const int b  = blockIdx.y;
  const int q0 = blockIdx.x * 16;
  const size_t bS = (size_t)b * SEQ;
  const size_t qRowBase = (size_t)b * NQ + q0;
  const size_t vTBase = (size_t)b * HIDDEN * SEQ;
  const int dBase = wave * 96;
  const float scale = 0.03608439182435161f;   // 1/sqrt(768)

  // stage the 16x768 Q tile once (contiguous 24 KB copy)
  for (int j = 0; j < 6; ++j) {
    const int c = tid + j * 256;             // chunk of 8 bf16
    const int row = c / 96, k0 = (c % 96) * 8;
    *reinterpret_cast<uint4*>(&qTile[row * HIDDEN + k0]) =
        *reinterpret_cast<const uint4*>(&qb[(qRowBase + row) * HIDDEN + k0]);
  }
  __syncthreads();

  v8f acc[6];
  for (int t = 0; t < 6; ++t) acc[t] = (v8f){};
  float mrow[8], lrow[8];
  for (int g = 0; g < 8; ++g) { mrow[g] = -1e30f; lrow[g] = 0.0f; }

  for (int s0 = 0; s0 < SEQ; s0 += 256) {
    const int sKey = s0 + wave * 32;         // this wave's private 32 keys
    if (s0 + 256 < SEQ) {                    // prefetch next 256-key K tile
      __builtin_prefetch(&kb[(bS + s0 + 256 + tid) * HIDDEN], 0, 0);
    }

    // ---- full-K scores for this wave's keys: 24 k-chunks x 2 n-tiles
    v8f sc0 = {}, sc1 = {};
    for (int kc = 0; kc < 24; ++kc) {
      const int kk = kc * 32;
      const v16bf aq  = loadFragA(&qTile[r * HIDDEN + kk], half);
      const v16bf bk0 = loadFragB(&kb[(bS + sKey + r) * HIDDEN + kk], half);
      const v16bf bk1 = loadFragB(&kb[(bS + sKey + 16 + r) * HIDDEN + kk], half);
      sc0 = wmma_bf16(aq, bk0, sc0);
      sc1 = wmma_bf16(aq, bk1, sc1);
    }

    // scale + mask bias; wave-local row max
    const float bias0 = (1.0f - mask[bS + sKey + r])      * -10000.0f;
    const float bias1 = (1.0f - mask[bS + sKey + 16 + r]) * -10000.0f;
    float v0[8], v1[8];
    for (int g = 0; g < 8; ++g) {
      v0[g] = sc0[g] * scale + bias0;
      v1[g] = sc1[g] * scale + bias1;
      float x = fmaxf(v0[g], v1[g]);     // row max over 16 lanes (xor < 16)
      for (int mk = 8; mk >= 1; mk >>= 1) x = fmaxf(x, __shfl_xor(x, mk, 32));
      if (r == 0) rowMaxB[g + half * 8][wave] = x;
    }
    __syncthreads();

    // global row max across waves; exp tiles; wave-local sums; publish P
    float esc[8];
    for (int g = 0; g < 8; ++g) {
      const int row = g + half * 8;
      const float4 ma = *reinterpret_cast<const float4*>(&rowMaxB[row][0]);
      const float4 mb = *reinterpret_cast<const float4*>(&rowMaxB[row][4]);
      const float mNew = fmaxf(mrow[g],
          fmaxf(fmaxf(fmaxf(ma.x, ma.y), fmaxf(ma.z, ma.w)),
                fmaxf(fmaxf(mb.x, mb.y), fmaxf(mb.z, mb.w))));
      const float e0 = __expf(v0[g] - mNew), e1 = __expf(v1[g] - mNew);
      float s = e0 + e1;                 // wave-local row sum over 16 lanes
      for (int mk = 8; mk >= 1; mk >>= 1) s += __shfl_xor(s, mk, 32);
      if (r == 0) rowSumB[row][wave] = s;
      esc[g]  = __expf(mrow[g] - mNew);
      mrow[g] = mNew;
      pBuf[row * 256 + (sKey - s0) + r]      = f32_bf16(e0);
      pBuf[row * 256 + (sKey - s0) + 16 + r] = f32_bf16(e1);
    }
    __syncthreads();

    // merge row sums, rescale accumulators
    for (int g = 0; g < 8; ++g) {
      const int row = g + half * 8;
      const float4 sa = *reinterpret_cast<const float4*>(&rowSumB[row][0]);
      const float4 sb = *reinterpret_cast<const float4*>(&rowSumB[row][4]);
      const float tot = (sa.x + sa.y + sa.z + sa.w) + (sb.x + sb.y + sb.z + sb.w);
      lrow[g] = lrow[g] * esc[g] + tot;
    }
    for (int t = 0; t < 6; ++t)
      for (int g = 0; g < 8; ++g) acc[t][g] *= esc[g];

    // ---- context: acc += P[16x256] @ V[256x96(this wave)], V^T from global
    for (int kc = 0; kc < 8; ++kc) {
      const v16bf pA = loadFragA(&pBuf[r * 256 + kc * 32], half);
      for (int t = 0; t < 6; ++t) {
        const int n = dBase + t * 16 + r;
        const v16bf bV = loadFragB(&vbT[vTBase + (size_t)n * SEQ + s0 + kc * 32], half);
        acc[t] = wmma_bf16(pA, bV, acc[t]);
      }
    }
    // next iteration's rowMaxB writes are fenced by its own barrier; pBuf
    // reads above complete before any thread passes that barrier.
  }

  // ---- epilogue: normalize by l and store f32 context
  for (int t = 0; t < 6; ++t) {
    const int n = dBase + t * 16 + r;
    for (int g = 0; g < 8; ++g) {
      const int row = g + half * 8;
      out[(qRowBase + row) * HIDDEN + n] = acc[t][g] / lrow[g];
    }
  }
}

// ---- host launch ---------------------------------------------------------

extern "C" void kernel_launch(void* const* d_in, const int* in_sizes, int n_in,
                              void* d_out, int out_size, void* d_ws, size_t ws_size,
                              hipStream_t stream) {
  const float* hidden  = (const float*)d_in[0];
  const float* queries = (const float*)d_in[1];
  const float* mask    = (const float*)d_in[2];
  const float* Wq      = (const float*)d_in[3];
  const float* bq      = (const float*)d_in[4];
  const float* Wk      = (const float*)d_in[5];
  const float* bk      = (const float*)d_in[6];
  const float* Wv      = (const float*)d_in[7];
  const float* bv      = (const float*)d_in[8];
  float* out = (float*)d_out;

  // workspace: bf16 q (6.3 MB) | bf16 k (50.3 MB) | bf16 V^T (50.3 MB)
  unsigned short* qb  = (unsigned short*)d_ws;
  unsigned short* kb  = qb + (size_t)BATCH * NQ  * HIDDEN;
  unsigned short* vbT = kb + (size_t)BATCH * SEQ * HIDDEN;

  const dim3 blk(256);
  proj_gemm_bf16<<<dim3(BATCH * NQ  / 64, HIDDEN / 64), blk, 0, stream>>>(queries, Wq, bq, qb,  0);
  proj_gemm_bf16<<<dim3(BATCH * SEQ / 64, HIDDEN / 64), blk, 0, stream>>>(hidden,  Wk, bk, kb,  0);
  proj_gemm_bf16<<<dim3(BATCH * SEQ / 64, HIDDEN / 64), blk, 0, stream>>>(hidden,  Wv, bv, vbT, 1);
  attn_flash_bf16<<<dim3(NQ / 16, BATCH), blk, 0, stream>>>(qb, kb, vbT, mask, out);
}
